// CAR_52785148068138
// MI455X (gfx1250) — compile-verified, matrix operand order checked by reading, not run on hardware
//
#include <hip/hip_runtime.h>
#include <hip/hip_bf16.h>
#include <math.h>

// ---------------------------------------------------------------------------
// CAR forward for MI455X (gfx1250, wave32, WMMA).
// Every conv = implicit GEMM on v_wmma_f32_16x16x32_f16:
//   A = f16 weights, pre-converted & padded to [Cout_pad16][K_pad32] -> one
//       aligned v16h load per k-step, no guards, no conversion math.
//   B = im2col gather (zero/reflect pad folded into the gather), f16 in-reg.
//   Each wave owns TWO 16-pixel tiles (NT=2): one A fragment feeds two WMMAs
//   with independent accumulators (ILP across the WMMA RAW hazard window).
// Epilogue fuses bias + eval-BN + {lrelu,relu,sigmoid} + residual*scale.
// DCNv2 fuses bilinear sampling * modulation into the B gather.
// All intermediates ping-pong in workspace slots (L2-resident, 192MB L2).
// ---------------------------------------------------------------------------

typedef __attribute__((ext_vector_type(16))) _Float16 v16h;
typedef __attribute__((ext_vector_type(8)))  float    v8f;

#define BN_EPS 1e-5f

__device__ __forceinline__ float sigmoidf_(float x) { return 1.f / (1.f + __expf(-x)); }

__device__ __forceinline__ int reflect_idx(int i, int n) {
  i = (i < 0) ? -i : i;
  i = (i >= n) ? (2 * n - 2 - i) : i;
  return i;
}

// C/D 16x16 f32 tile layout: vgpr r, lane l -> M = r + 8*(l>>4), N = l&15.
__device__ __forceinline__ void wmma_epilogue(
    const v8f& acc, int lane, int mt, int Cout, int p0, int npix,
    const float* __restrict__ bias,
    const float* __restrict__ bng, const float* __restrict__ bnb,
    const float* __restrict__ bnm, const float* __restrict__ bnv,
    const float* __restrict__ resid, float res_scale, int act,
    float* __restrict__ Y) {
  const int khalf = lane >> 4;
  const int pix   = p0 + (lane & 15);
#pragma unroll
  for (int r = 0; r < 8; ++r) {
    const int row = mt * 16 + r + 8 * khalf;
    if (row < Cout) {
      float v = acc[r];
      if (bias) v += bias[row];
      if (bng) {
        const float sc = bng[row] * rsqrtf(bnv[row] + BN_EPS);
        v = (v - bnm[row]) * sc + bnb[row];
      }
      if      (act == 1) v = (v >= 0.f) ? v : 0.2f * v;   // lrelu(0.2)
      else if (act == 2) v = fmaxf(v, 0.f);               // relu
      else if (act == 3) v = sigmoidf_(v);                // sigmoid
      v *= res_scale;
      if (resid) v += resid[(size_t)row * npix + pix];
      Y[(size_t)row * npix + pix] = v;
    }
  }
}

// Weight convert/pad: f32 [Cout][K] -> f16 [CoutP][Kp], zero-filled padding.
__global__ void k_cvt_w16(const float* __restrict__ src, _Float16* __restrict__ dst,
                          int Cout, int K, int CoutP, int Kp) {
  const int n = CoutP * Kp;
  const int i = blockIdx.x * blockDim.x + threadIdx.x;
  if (i >= n) return;
  const int k = i % Kp, r = i / Kp;
  dst[i] = (_Float16)((r < Cout && k < K) ? src[(size_t)r * K + k] : 0.f);
}

// Generic stride-1 "same" conv. KS/PADMODE compile-time; pad = KS/2.
// Block = 4 waves, each wave: 16 out-ch x 32 pixels. npix % 128 == 0 always.
template <int KS, int PADMODE>
__global__ void __launch_bounds__(128)
k_conv_wmma(const float* __restrict__ X, const _Float16* __restrict__ Wt,
            const float* __restrict__ bias,
            const float* __restrict__ bng, const float* __restrict__ bnb,
            const float* __restrict__ bnm, const float* __restrict__ bnv,
            const float* __restrict__ resid, float res_scale,
            float* __restrict__ Y,
            int C, int H, int Wd, int Cout, int Kp, int act) {
  constexpr int PAD = KS / 2;
  constexpr int KK2 = KS * KS;
  const int lane  = threadIdx.x & 31;
  const int wave  = threadIdx.x >> 5;
  const int npix  = H * Wd;
  const int p0    = (blockIdx.x * 4 + wave) * 32;   // 32 pixels per wave
  const int mt    = blockIdx.y;

  const int khalf = lane >> 4;
  const int ln    = lane & 15;
  const int pix0  = p0 + ln;
  const int pix1  = p0 + 16 + ln;
  const int oh0 = pix0 / Wd, ow0 = pix0 - oh0 * Wd;
  const int oh1 = pix1 / Wd, ow1 = pix1 - oh1 * Wd;
  const _Float16* __restrict__ Arow = Wt + (size_t)(mt * 16 + ln) * Kp;

  v8f acc0 = {}, acc1 = {};
  for (int k0 = 0; k0 < Kp; k0 += 32) {
    if (k0 + 32 < Kp)                      // gfx1250 global_prefetch_b8
      __builtin_prefetch(Arow + k0 + 32, 0, 1);
    const v16h a = *(const v16h*)(Arow + k0 + khalf * 16);
    v16h b0, b1;
    if (KS == 1) {
      int k = k0 + khalf * 16;
#pragma unroll
      for (int j = 0; j < 16; ++j, ++k) {
        const bool ok = (k < C);
        b0[j] = (_Float16)(ok ? X[(size_t)k * npix + pix0] : 0.f);
        b1[j] = (_Float16)(ok ? X[(size_t)k * npix + pix1] : 0.f);
      }
    } else {
      int k  = k0 + khalf * 16;
      int ci = k / KK2;                    // constant divisor (KS compile-time)
      int rem = k - ci * KK2;
      int kr = rem / KS, kc = rem - kr * KS;
#pragma unroll
      for (int j = 0; j < 16; ++j) {
        float x0 = 0.f, x1 = 0.f;
        if (ci < C) {
          const float* __restrict__ Xc = X + (size_t)ci * npix;
          int ih = oh0 + kr - PAD, iw = ow0 + kc - PAD;
          if (PADMODE == 1) {
            x0 = Xc[reflect_idx(ih, H) * Wd + reflect_idx(iw, Wd)];
          } else if (ih >= 0 && ih < H && iw >= 0 && iw < Wd) {
            x0 = Xc[ih * Wd + iw];
          }
          ih = oh1 + kr - PAD; iw = ow1 + kc - PAD;
          if (PADMODE == 1) {
            x1 = Xc[reflect_idx(ih, H) * Wd + reflect_idx(iw, Wd)];
          } else if (ih >= 0 && ih < H && iw >= 0 && iw < Wd) {
            x1 = Xc[ih * Wd + iw];
          }
        }
        b0[j] = (_Float16)x0;
        b1[j] = (_Float16)x1;
        if (++kc == KS) { kc = 0; if (++kr == KS) { kr = 0; ++ci; } }
      }
    }
    acc0 = __builtin_amdgcn_wmma_f32_16x16x32_f16(false, a, false, b0,
                                                  (short)0, acc0, false, false);
    acc1 = __builtin_amdgcn_wmma_f32_16x16x32_f16(false, a, false, b1,
                                                  (short)0, acc1, false, false);
  }
  wmma_epilogue(acc0, lane, mt, Cout, p0, npix,
                bias, bng, bnb, bnm, bnv, resid, res_scale, act, Y);
  wmma_epilogue(acc1, lane, mt, Cout, p0 + 16, npix,
                bias, bng, bnb, bnm, bnv, resid, res_scale, act, Y);
}

// DCNv2 B-gather: bilinear sample of zero-padded X at offset position * mask.
__device__ __forceinline__ float dcn_sample(
    const float* __restrict__ Xc, const float* __restrict__ OFF,
    const float* __restrict__ MOD, int n9, int kr, int kc,
    int oh, int ow, int pix, int H, int Wd, int npix) {
  const float orow = OFF[(size_t)n9 * npix + pix];
  const float ocol = OFF[(size_t)(9 + n9) * npix + pix];
  const float mod  = MOD[(size_t)n9 * npix + pix];
  const float Hp = (float)(H + 2), Wp = (float)(Wd + 2);
  float px = (float)(oh + kr) + orow;        // padded coords: (oh+1)+(kr-1)
  float py = (float)(ow + kc) + ocol;
  px = fminf(fmaxf(px, 0.f), Hp - 1.f);
  py = fminf(fmaxf(py, 0.f), Wp - 1.f);
  const float x0 = fminf(floorf(px), Hp - 1.f);
  const float y0 = fminf(floorf(py), Wp - 1.f);
  const float x1 = fminf(x0 + 1.f, Hp - 1.f);
  const float y1 = fminf(y0 + 1.f, Wp - 1.f);
  const float glt = (1.f + (x0 - px)) * (1.f + (y0 - py));
  const float grb = (1.f - (x1 - px)) * (1.f - (y1 - py));
  const float glb = (1.f + (x0 - px)) * (1.f - (y1 - py));
  const float grt = (1.f - (x1 - px)) * (1.f + (y0 - py));
  const int ix0 = (int)x0 - 1, iy0 = (int)y0 - 1;   // back to unpadded
  const int ix1 = (int)x1 - 1, iy1 = (int)y1 - 1;
  float s00 = 0.f, s11 = 0.f, s01 = 0.f, s10 = 0.f;
  if (ix0 >= 0 && ix0 < H && iy0 >= 0 && iy0 < Wd) s00 = Xc[ix0 * Wd + iy0];
  if (ix1 >= 0 && ix1 < H && iy1 >= 0 && iy1 < Wd) s11 = Xc[ix1 * Wd + iy1];
  if (ix0 >= 0 && ix0 < H && iy1 >= 0 && iy1 < Wd) s01 = Xc[ix0 * Wd + iy1];
  if (ix1 >= 0 && ix1 < H && iy0 >= 0 && iy0 < Wd) s10 = Xc[ix1 * Wd + iy0];
  return mod * (glt * s00 + grb * s11 + glb * s01 + grt * s10);
}

__global__ void __launch_bounds__(128)
k_deform_wmma(const float* __restrict__ X, const float* __restrict__ OFF,
              const float* __restrict__ MOD, const _Float16* __restrict__ Wt,
              const float* __restrict__ bias, float* __restrict__ Y,
              int C, int H, int Wd, int Cout, int Kp) {
  const int lane  = threadIdx.x & 31;
  const int wave  = threadIdx.x >> 5;
  const int npix  = H * Wd;
  const int p0    = (blockIdx.x * 4 + wave) * 32;
  const int mt    = blockIdx.y;

  const int khalf = lane >> 4;
  const int ln    = lane & 15;
  const int pix0  = p0 + ln;
  const int pix1  = p0 + 16 + ln;
  const int oh0 = pix0 / Wd, ow0 = pix0 - oh0 * Wd;
  const int oh1 = pix1 / Wd, ow1 = pix1 - oh1 * Wd;
  const _Float16* __restrict__ Arow = Wt + (size_t)(mt * 16 + ln) * Kp;

  v8f acc0 = {}, acc1 = {};
  for (int k0 = 0; k0 < Kp; k0 += 32) {
    const v16h a = *(const v16h*)(Arow + k0 + khalf * 16);
    v16h b0, b1;
    int k  = k0 + khalf * 16;
    int ci = k / 9;
    int rem = k - ci * 9;
    int kr = rem / 3, kc = rem - kr * 3;
#pragma unroll
    for (int j = 0; j < 16; ++j) {
      float x0 = 0.f, x1 = 0.f;
      if (ci < C) {
        const float* Xc = X + (size_t)ci * npix;
        const int n9 = kr * 3 + kc;
        x0 = dcn_sample(Xc, OFF, MOD, n9, kr, kc, oh0, ow0, pix0, H, Wd, npix);
        x1 = dcn_sample(Xc, OFF, MOD, n9, kr, kc, oh1, ow1, pix1, H, Wd, npix);
      }
      b0[j] = (_Float16)x0;
      b1[j] = (_Float16)x1;
      if (++kc == 3) { kc = 0; if (++kr == 3) { kr = 0; ++ci; } }
    }
    acc0 = __builtin_amdgcn_wmma_f32_16x16x32_f16(false, a, false, b0,
                                                  (short)0, acc0, false, false);
    acc1 = __builtin_amdgcn_wmma_f32_16x16x32_f16(false, a, false, b1,
                                                  (short)0, acc1, false, false);
  }
  wmma_epilogue(acc0, lane, mt, Cout, p0, npix,
                bias, nullptr, nullptr, nullptr, nullptr, nullptr, 1.f, 0, Y);
  wmma_epilogue(acc1, lane, mt, Cout, p0 + 16, npix,
                bias, nullptr, nullptr, nullptr, nullptr, nullptr, 1.f, 0, Y);
}

// ----------------------------- elementwise ---------------------------------

__global__ void k_addcoords(const float* __restrict__ X, float* __restrict__ Y,
                            int C, int H, int W) {
  const int n = (C + 3) * H * W;
  const int i = blockIdx.x * blockDim.x + threadIdx.x;
  if (i >= n) return;
  const int w = i % W;
  const int h = (i / W) % H;
  const int c = i / (W * H);
  const float xx = (W > 1) ? (-1.f + 2.f * (float)w / (float)(W - 1)) : 0.f;
  const float yy = (H > 1) ? (-1.f + 2.f * (float)h / (float)(H - 1)) : 0.f;
  float v;
  if      (c <  C)     v = X[i];
  else if (c == C)     v = xx;
  else if (c == C + 1) v = yy;
  else                 v = sqrtf((xx - 0.5f) * (xx - 0.5f) + (yy - 0.5f) * (yy - 0.5f));
  Y[i] = v;
}

__global__ void k_unshuffle2(const float* __restrict__ X, float* __restrict__ Y,
                             int C, int H, int W) {   // (C,H,W) -> (4C,H/2,W/2)
  const int Ho = H >> 1, Wo = W >> 1;
  const int n = C * 4 * Ho * Wo;
  const int i = blockIdx.x * blockDim.x + threadIdx.x;
  if (i >= n) return;
  const int wo = i % Wo;
  const int ho = (i / Wo) % Ho;
  const int co = i / (Wo * Ho);
  const int c = co >> 2, sy = (co >> 1) & 1, sx = co & 1;
  Y[i] = X[((size_t)c * H + (ho * 2 + sy)) * W + (wo * 2 + sx)];
}

__global__ void k_shuffle2(const float* __restrict__ X, float* __restrict__ Y,
                           int C4, int H, int W) {    // (4C,H,W) -> (C,2H,2W)
  const int C = C4 >> 2;
  const int Ho = H << 1, Wo = W << 1;
  const int n = C * Ho * Wo;
  const int i = blockIdx.x * blockDim.x + threadIdx.x;
  if (i >= n) return;
  const int wo = i % Wo;
  const int ho = (i / Wo) % Ho;
  const int c = i / (Wo * Ho);
  const int ry = ho & 1, rx = wo & 1;
  Y[i] = X[((size_t)(c * 4 + ry * 2 + rx) * H + (ho >> 1)) * W + (wo >> 1)];
}

__global__ void k_bilm(const float* __restrict__ X, float* __restrict__ Y,
                       int C, int H, int W) {
  const int n = C * H * W;
  const int i = blockIdx.x * blockDim.x + threadIdx.x;
  if (i >= n) return;
  const int w = i % W;
  const int h = (i / W) % H;
  const int c = i / (W * H);
  float mx = -1e30f, mn = -1e30f;
  for (int dh = -1; dh <= 1; ++dh)
    for (int dw = -1; dw <= 1; ++dw) {
      const int hh = h + dh, ww = w + dw;
      if (hh < 0 || hh >= H || ww < 0 || ww >= W) continue;
      const float s = sigmoidf_(X[((size_t)c * H + hh) * W + ww]);
      mx = fmaxf(mx, s);
      mn = fmaxf(mn, -s);
    }
  Y[i] = X[i] * (mx + mn);
}

__global__ void k_add(const float* __restrict__ A, const float* __restrict__ B,
                      float* __restrict__ Y, int n) {
  const int i = blockIdx.x * blockDim.x + threadIdx.x;
  if (i < n) Y[i] = A[i] + B[i];
}

// ------------------------------- host side ---------------------------------

static inline int ceil16(int x) { return (x + 15) & ~15; }
static inline int ceil32(int x) { return (x + 31) & ~31; }

static inline void conv(hipStream_t st, const float* X, const _Float16* W16,
                        const float* bias, const float* bng, const float* bnb,
                        const float* bnm, const float* bnv,
                        const float* resid, float rs, float* Y,
                        int C, int H, int W, int Cout, int ks, int padmode, int act) {
  const int npix = H * W;
  const int Kp = ceil32(C * ks * ks);
  dim3 grid(npix / 128, ceil16(Cout) / 16, 1);
  dim3 blk(128, 1, 1);
  switch (ks * 10 + padmode) {
    case 10:
      k_conv_wmma<1, 0><<<grid, blk, 0, st>>>(X, W16, bias, bng, bnb, bnm, bnv,
                                              resid, rs, Y, C, H, W, Cout, Kp, act);
      break;
    case 30:
      k_conv_wmma<3, 0><<<grid, blk, 0, st>>>(X, W16, bias, bng, bnb, bnm, bnv,
                                              resid, rs, Y, C, H, W, Cout, Kp, act);
      break;
    case 31:
      k_conv_wmma<3, 1><<<grid, blk, 0, st>>>(X, W16, bias, bng, bnb, bnm, bnv,
                                              resid, rs, Y, C, H, W, Cout, Kp, act);
      break;
    default:
      k_conv_wmma<5, 1><<<grid, blk, 0, st>>>(X, W16, bias, bng, bnb, bnm, bnv,
                                              resid, rs, Y, C, H, W, Cout, Kp, act);
      break;
  }
}

#define EW(kname, n, ...) kname<<<((n) + 255) / 256, 256, 0, stream>>>(__VA_ARGS__)

// d_in layout: x, then params in make_params() dict order (BN tuples expand to 4).
enum {
  I_X = 0,
  I_CC_W, I_CC_B,
  I_DC_PW, I_DC_PB, I_DC_MW, I_DC_MB, I_DC_W, I_DC_B,
  I_DS1_W, I_DS1_B, I_DS1_G, I_DS1_BE, I_DS1_M, I_DS1_V,
  I_DS2_W, I_DS2_B, I_DS2_G, I_DS2_BE, I_DS2_M, I_DS2_V,
  I_DS4_W, I_DS4_B, I_DS4_G, I_DS4_BE, I_DS4_M, I_DS4_V,
  I_R4_W1, I_R4_B1, I_R4_W2, I_R4_B2,
  I_DS8_W, I_DS8_B, I_DS8_G, I_DS8_BE, I_DS8_M, I_DS8_V,
  I_CC1_W, I_CC1_B,
  I_HEAD_W, I_HEAD_B,
  I_BD_W1, I_BD_B1, I_BD_W2, I_BD_B2,
  I_BT_W, I_BT_B,
  I_UP_W, I_UP_B,
  I_TAIL_W, I_TAIL_B,
  I_CC2_W, I_CC2_B,
  I_O1_W, I_O1_B, I_O1_G, I_O1_BE, I_O1_M, I_O1_V,
  I_O2_W, I_O2_B
};

extern "C" void kernel_launch(void* const* d_in, const int* in_sizes, int n_in,
                              void* d_out, int out_size, void* d_ws, size_t ws_size,
                              hipStream_t stream) {
  (void)in_sizes; (void)out_size; (void)ws_size;
  const float* P[64];
  for (int i = 0; i < 61 && i < n_in; ++i) P[i] = (const float*)d_in[i];
  float* OUT = (float*)d_out;

  // workspace slots (all intermediates stay L2-resident on the 192MB L2)
  char* ws = (char*)d_ws;
  float* S0 = (float*)(ws);                                  // 34 MB
  float* S1 = (float*)(ws + (size_t)34 * 1024 * 1024);       // 34 MB
  float* S2 = (float*)(ws + (size_t)68 * 1024 * 1024);       // 34 MB
  float* S3 = (float*)(ws + (size_t)102 * 1024 * 1024);      // 10 MB
  float* S4 = (float*)(ws + (size_t)112 * 1024 * 1024);      //  6 MB

  // ---- convert all conv weights to padded f16 tiles (bump region @118MB) ---
  _Float16* wbump = (_Float16*)(ws + (size_t)118 * 1024 * 1024);
  auto cvt = [&](const float* src, int Cout, int K) -> const _Float16* {
    const int CoutP = ceil16(Cout), Kp = ceil32(K);
    const int n = CoutP * Kp;
    k_cvt_w16<<<(n + 255) / 256, 256, 0, stream>>>(src, wbump, Cout, K, CoutP, Kp);
    const _Float16* r = wbump;
    wbump += n;
    return r;
  };
  const _Float16* Wcc   = cvt(P[I_CC_W],   64,   9);
  const _Float16* Wdcp  = cvt(P[I_DC_PW],  18, 576);
  const _Float16* Wdcm  = cvt(P[I_DC_MW],   9, 576);
  const _Float16* Wdc   = cvt(P[I_DC_W],   64, 576);
  const _Float16* Wds1  = cvt(P[I_DS1_W],  64, 1600);
  const _Float16* Wds2  = cvt(P[I_DS2_W], 128, 256);
  const _Float16* Wds4  = cvt(P[I_DS4_W], 128, 512);
  const _Float16* Wds8  = cvt(P[I_DS8_W], 256, 512);
  const _Float16* Wr4a[5], *Wr4b[5];
  for (int i = 0; i < 5; ++i) {
    Wr4a[i] = cvt(P[I_R4_W1] + (size_t)i * 128 * 1152, 128, 1152);
    Wr4b[i] = cvt(P[I_R4_W2] + (size_t)i * 128 * 1152, 128, 1152);
  }
  const _Float16* Wcc1  = cvt(P[I_CC1_W],   3, 259);
  const _Float16* Whead = cvt(P[I_HEAD_W], 64,  27);
  const _Float16* Wbd1[16], *Wbd2[16];
  for (int i = 0; i < 16; ++i) {
    Wbd1[i] = cvt(P[I_BD_W1] + (size_t)i * 64 * 576, 64, 576);
    Wbd2[i] = cvt(P[I_BD_W2] + (size_t)i * 64 * 576, 64, 576);
  }
  const _Float16* Wbt   = cvt(P[I_BT_W],   64, 576);
  const _Float16* Wup[3];
  for (int i = 0; i < 3; ++i)
    Wup[i] = cvt(P[I_UP_W] + (size_t)i * 256 * 576, 256, 576);
  const _Float16* Wtail = cvt(P[I_TAIL_W],  3, 576);
  const _Float16* Wcc2  = cvt(P[I_CC2_W],   3,   6);
  const _Float16* Wo1   = cvt(P[I_O1_W],    3,   3);
  const _Float16* Wo2   = cvt(P[I_O2_W],    1,   3);

  const int H = 192, W = 640;          // input spatial
  const int NP = H * W;                // 122880

  // ---------------- DSN ----------------
  EW(k_addcoords, 9 * NP, P[I_X], S3, 6, H, W);                       // 6 -> 9ch
  conv(stream, S3, Wcc, P[I_CC_B], 0, 0, 0, 0, 0, 1.f, S0,
       9, H, W, 64, 1, 0, 0);                                         // cc 9->64
  conv(stream, S0, Wdcp, P[I_DC_PB], 0, 0, 0, 0, 0, 1.f, S3,
       64, H, W, 18, 3, 0, 0);                                        // offsets
  conv(stream, S0, Wdcm, P[I_DC_MB], 0, 0, 0, 0, 0, 1.f, S4,
       64, H, W, 9, 3, 0, 3);                                         // mask+sig
  {
    dim3 grid(NP / 128, 4, 1);
    k_deform_wmma<<<grid, dim3(128, 1, 1), 0, stream>>>(
        S0, S3, S4, Wdc, P[I_DC_B], S1, 64, H, W, 64, 576);           // DCNv2
  }
  conv(stream, S1, Wds1, P[I_DS1_B], P[I_DS1_G], P[I_DS1_BE],
       P[I_DS1_M], P[I_DS1_V], 0, 1.f, S0, 64, H, W, 64, 5, 1, 1);    // 5x5 refl+bn+lrelu
  EW(k_unshuffle2, 256 * (H / 2) * (W / 2), S0, S1, 64, H, W);
  conv(stream, S1, Wds2, P[I_DS2_B], P[I_DS2_G], P[I_DS2_BE],
       P[I_DS2_M], P[I_DS2_V], 0, 1.f, S0, 256, H / 2, W / 2, 128, 1, 0, 0);
  EW(k_unshuffle2, 512 * (H / 4) * (W / 4), S0, S1, 128, H / 2, W / 2);
  conv(stream, S1, Wds4, P[I_DS4_B], P[I_DS4_G], P[I_DS4_BE],
       P[I_DS4_M], P[I_DS4_V], 0, 1.f, S2, 512, H / 4, W / 4, 128, 1, 0, 0);
  {   // 5 residual blocks @ 128ch, 48x160, reflect pad
    const int h4 = H / 4, w4 = W / 4;
    const float* rin = S2;
    for (int i = 0; i < 5; ++i) {
      float* rout = (i & 1) ? S3 : S0;
      conv(stream, rin, Wr4a[i], P[I_R4_B1] + i * 128,
           0, 0, 0, 0, 0, 1.f, S1, 128, h4, w4, 128, 3, 1, 1);        // conv+lrelu
      conv(stream, S1, Wr4b[i], P[I_R4_B2] + i * 128,
           0, 0, 0, 0, rin, 1.f, rout, 128, h4, w4, 128, 3, 1, 0);    // conv + r
      rin = rout;
    }
    EW(k_add, 128 * h4 * w4, S2, rin, S1, 128 * h4 * w4);             // h = h + r
  }
  EW(k_unshuffle2, 512 * (H / 8) * (W / 8), S1, S0, 128, H / 4, W / 4);
  conv(stream, S0, Wds8, P[I_DS8_B], P[I_DS8_G], P[I_DS8_BE],
       P[I_DS8_M], P[I_DS8_V], 0, 1.f, S1, 512, H / 8, W / 8, 256, 1, 0, 0);
  EW(k_bilm, 256 * (H / 8) * (W / 8), S1, S0, 256, H / 8, W / 8);     // BILM

  // ---------------- EDSR ----------------
  const int h8 = H / 8, w8 = W / 8;                                   // 24 x 80
  EW(k_addcoords, 259 * h8 * w8, S0, S1, 256, h8, w8);
  conv(stream, S1, Wcc1, P[I_CC1_B], 0, 0, 0, 0, 0, 1.f, S3,
       259, h8, w8, 3, 1, 0, 0);                                      // cc1 259->3
  conv(stream, S3, Whead, P[I_HEAD_B], 0, 0, 0, 0, 0, 1.f, S2,
       3, h8, w8, 64, 3, 0, 0);                                       // head
  {   // 16 resblocks @ 64ch, res_scale = 0.1
    const float* rin = S2;
    for (int i = 0; i < 16; ++i) {
      float* rout = (i & 1) ? S3 : S0;
      conv(stream, rin, Wbd1[i], P[I_BD_B1] + i * 64,
           0, 0, 0, 0, 0, 1.f, S1, 64, h8, w8, 64, 3, 0, 2);          // conv+relu
      conv(stream, S1, Wbd2[i], P[I_BD_B2] + i * 64,
           0, 0, 0, 0, rin, 0.1f, rout, 64, h8, w8, 64, 3, 0, 0);     // r + 0.1*t
      rin = rout;
    }
    conv(stream, rin, Wbt, P[I_BT_B], 0, 0, 0, 0, S2, 1.f, S0,
         64, h8, w8, 64, 3, 0, 0);                                    // h + conv(r)
  }
  {   // Upsampler x8: 3x (conv 64->256, PixelShuffle(2))
    int hh = h8, wu = w8;
    for (int i = 0; i < 3; ++i) {
      conv(stream, S0, Wup[i], P[I_UP_B] + i * 256,
           0, 0, 0, 0, 0, 1.f, S1, 64, hh, wu, 256, 3, 0, 0);
      EW(k_shuffle2, 64 * (hh * 2) * (wu * 2), S1, S0, 256, hh, wu);
      hh *= 2; wu *= 2;
    }
  }
  conv(stream, S0, Wtail, P[I_TAIL_B], 0, 0, 0, 0, 0, 1.f, S1,
       64, H, W, 3, 3, 0, 0);                                         // tail 64->3
  EW(k_addcoords, 6 * NP, S1, S0, 3, H, W);
  conv(stream, S0, Wcc2, P[I_CC2_B], 0, 0, 0, 0, 0, 1.f, S1,
       6, H, W, 3, 1, 0, 0);                                          // cc2 6->3
  conv(stream, S1, Wo1, P[I_O1_B], P[I_O1_G], P[I_O1_BE],
       P[I_O1_M], P[I_O1_V], 0, 1.f, S0, 3, H, W, 3, 1, 0, 2);        // o1+bn+relu
  conv(stream, S0, Wo2, P[I_O2_B], 0, 0, 0, 0, 0, 1.f, OUT,
       3, H, W, 1, 1, 0, 3);                                          // o2+sigmoid
}
